// QuantLlamaAttentionFused_30837865185837
// MI455X (gfx1250) — matrix operand order repaired and checked by reading, not compiled, not run on hardware
//
#include <hip/hip_runtime.h>
#include <stdint.h>

// ---------------------------------------------------------------------------
// Problem constants (match reference)
// ---------------------------------------------------------------------------
#define BB   2
#define SS   1024
#define HH   4096
#define NH   32
#define NKV  8
#define HD   128
#define NQKV (NH + 2 * NKV)          // 48 heads in fused qkv
#define QKVN (NQKV * HD)             // 6144

typedef __bf16 v16bf __attribute__((ext_vector_type(16)));
typedef float  v8f   __attribute__((ext_vector_type(8)));

union FragU { uint4 u[2]; v16bf v; };   // 32 bytes, 16-byte aligned

// ---------------------------------------------------------------------------
// Tensor Data Mover support (CDNA5).  D# built per CDNA5 ISA
// ---------------------------------------------------------------------------
#if defined(__has_builtin)
#  if __has_builtin(__builtin_amdgcn_tensor_load_to_lds)
#    define TDM_OK 1
#  endif
#endif
#ifndef TDM_OK
#  define TDM_OK 0
#endif

// TDM writes LDS behind the compiler's back.  Two pieces are needed so LLVM
// cannot fold the subsequent ds_loads to undef:
//  1) lds_escape(): capture the __shared__ base pointer in an asm operand so
//     the object is "escaped" (a may-write asm can now reach it);
//  2) memory clobbers at every wait/barrier so loads are re-issued afterwards.
__device__ __forceinline__ void lds_escape(void* p) {
  asm volatile("" : : "v"(p) : "memory");
}
__device__ __forceinline__ void mem_clobber() { asm volatile("" ::: "memory"); }
__device__ __forceinline__ void wg_sync() { __syncthreads(); mem_clobber(); }

#if TDM_OK
typedef unsigned int u32x4 __attribute__((ext_vector_type(4)));
typedef int          i32x4 __attribute__((ext_vector_type(4)));
typedef int          i32x8 __attribute__((ext_vector_type(8)));

// 2D tile DMA: global (2-byte elements, row stride stride0) -> LDS with row
// padding.  pad codes: interval c -> pad after 8<<c bytes; amount a -> (a+1) DW.
__device__ __forceinline__ void tdm_load_2d(uint32_t lds_off, const void* gptr,
                                            uint64_t stride0_elems,
                                            uint32_t tile_w, uint32_t tile_h,
                                            uint32_t pad_int_c, uint32_t pad_amt_c) {
  uint64_t ga = (uint64_t)(uintptr_t)gptr;
  u32x4 g0;
  g0[0] = 1u;                                            // count=1, user mode
  g0[1] = lds_off;                                       // lds_addr (bytes)
  g0[2] = (uint32_t)ga;                                  // global_addr[31:0]
  g0[3] = (uint32_t)((ga >> 32) & 0x01FFFFFFull) | 0x80000000u; // [56:32] | type=2
  i32x8 g1;
  g1[0] = (int)((1u << 16)                               // data_size = 2 bytes
              | (1u << 20)                               // pad_enable
              | (pad_int_c << 22) | (pad_amt_c << 25));
  g1[1] = (int)((tile_w & 0xFFFFu) << 16);               // tensor_dim0 lo (=tile_w)
  g1[2] = (int)(((tile_w >> 16) & 0xFFFFu) | ((tile_h & 0xFFFFu) << 16)); // dim0 hi | dim1 lo
  g1[3] = (int)(((tile_h >> 16) & 0xFFFFu) | ((tile_w & 0xFFFFu) << 16)); // dim1 hi | tile_dim0
  g1[4] = (int)(tile_h & 0xFFFFu);                       // tile_dim1 | tile_dim2=0
  g1[5] = (int)(uint32_t)(stride0_elems & 0xFFFFFFFFull);// tensor_dim0_stride lo
  g1[6] = (int)(uint32_t)((stride0_elems >> 32) & 0xFFFFull); // stride hi | dim1_stride lo
  g1[7] = 0;
  i32x4 gz = {0, 0, 0, 0};
#if __clang_major__ >= 23
  i32x8 gz8 = {0, 0, 0, 0, 0, 0, 0, 0};
  __builtin_amdgcn_tensor_load_to_lds(g0, g1, gz, gz, gz8, 0);
#else
  __builtin_amdgcn_tensor_load_to_lds(g0, g1, gz, gz, 0);
#endif
}
__device__ __forceinline__ void tdm_wait0() {
  __builtin_amdgcn_s_wait_tensorcnt((short)0);
  mem_clobber();   // LDS contents changed by DMA: force reloads after this
}
#endif  // TDM_OK

__device__ __forceinline__ unsigned short f2bf(float f) {
  union { float f; uint32_t u; } x; x.f = f;
  uint32_t r = x.u + 0x7FFFu + ((x.u >> 16) & 1u);   // round-to-nearest-even
  return (unsigned short)(r >> 16);
}

__device__ __forceinline__ float rmax16(float v) {
  #pragma unroll
  for (int m = 1; m < 16; m <<= 1) v = fmaxf(v, __shfl_xor(v, m, 32));
  return v;  // masks < 16 stay within each 16-lane half of the wave32
}
__device__ __forceinline__ float rsum16(float v) {
  #pragma unroll
  for (int m = 1; m < 16; m <<= 1) v += __shfl_xor(v, m, 32);
  return v;
}

// ---------------------------------------------------------------------------
// fp32 -> bf16 conversion (activations)
// ---------------------------------------------------------------------------
__global__ void cvt_f32_bf16(const float* __restrict__ in,
                             unsigned short* __restrict__ out, int n) {
  int i = blockIdx.x * blockDim.x + threadIdx.x;
  if (i < n) out[i] = f2bf(in[i]);
}

// ---------------------------------------------------------------------------
// Fused transpose + fp32->bf16 for weights: in f32 [K,N] -> out bf16 [N,K].
// 32x32 LDS tile, 256 threads (32 wide x 8), coalesced both sides.
// ---------------------------------------------------------------------------
__global__ __launch_bounds__(256) void transpose_cvt(const float* __restrict__ in,
                                                     unsigned short* __restrict__ out,
                                                     int K, int N) {
  __shared__ float tile[32][33];
  int kb = blockIdx.y * 32, nb = blockIdx.x * 32;
  int tx = threadIdx.x & 31, ty = threadIdx.x >> 5;
  #pragma unroll
  for (int i = 0; i < 32; i += 8)
    tile[ty + i][tx] = in[(size_t)(kb + ty + i) * N + nb + tx];
  __syncthreads();
  #pragma unroll
  for (int i = 0; i < 32; i += 8)
    out[(size_t)(nb + ty + i) * K + kb + tx] = f2bf(tile[tx][ty + i]);
}

// ---------------------------------------------------------------------------
// Tiled bf16 WMMA GEMM:  C(f32)[M,N] = A_bf16[M,K] @ Wt_bf16[N,K]^T
// (Wt is the pre-transposed weight, so both LDS tiles are plain row copies.)
// Block: 256 threads (8 waves), tile 128M x 128N, K-step 32.
// TDM double-buffers the two tiles; DMA of tile k+1 overlaps WMMA on tile k.
// ---------------------------------------------------------------------------
__global__ __launch_bounds__(256) void gemm_bf16_wmma(
    const unsigned short* __restrict__ A, const unsigned short* __restrict__ Wt,
    float* __restrict__ C, int M, int N, int K) {
  __shared__ char smem[2 * 20480];       // per buffer: A 128x80B, Bt 128x80B
  lds_escape(smem);                      // TDM writes smem: make it visible

  const int t    = threadIdx.x;
  const int lane = t & 31;
  const int w    = t >> 5;
  const int half = lane >> 4;
  const int l16  = lane & 15;
  const int mBase = blockIdx.y * 128;
  const int nBase = blockIdx.x * 128;

  const v8f vzero = {0.f, 0.f, 0.f, 0.f, 0.f, 0.f, 0.f, 0.f};
  v8f acc[8];
  #pragma unroll
  for (int c = 0; c < 8; ++c) acc[c] = vzero;

#if TDM_OK
  if (t < 32) {   // wave 0 issues DMA; EXEC ignored by TDM, one issue per wave
    tdm_load_2d(0,     A  + (size_t)mBase * K, K, 32, 128, 3, 3);
    tdm_load_2d(10240, Wt + (size_t)nBase * K, K, 32, 128, 3, 3);
    tdm_wait0();
  }
  wg_sync();
#endif

  int buf = 0;
  for (int kb = 0; kb < K; kb += 32) {
    const char* sA = smem + buf * 20480;
    const char* sB = sA + 10240;
    const int nxt = buf ^ 1;

#if TDM_OK
    if (kb + 32 < K && t < 32) {   // prefetch next K tile into other buffer
      tdm_load_2d((uint32_t)(nxt * 20480),
                  A + (size_t)mBase * K + kb + 32, K, 32, 128, 3, 3);
      tdm_load_2d((uint32_t)(nxt * 20480 + 10240),
                  Wt + (size_t)nBase * K + kb + 32, K, 32, 128, 3, 3);
    }
#else
    {  // manual staging (fallback): both tiles are 128 rows x 32 bf16
      int row = t >> 1, ch = t & 1;
      const uint4* ga = (const uint4*)(A + (size_t)(mBase + row) * K + kb + ch * 16);
      uint4* sa = (uint4*)((char*)sA + row * 80 + ch * 32);
      sa[0] = ga[0]; sa[1] = ga[1];
      const uint4* gb = (const uint4*)(Wt + (size_t)(nBase + row) * K + kb + ch * 16);
      uint4* sb = (uint4*)((char*)sB + row * 80 + ch * 32);
      sb[0] = gb[0]; sb[1] = gb[1];
    }
    wg_sync();
#endif

    // A fragment: lane holds row (w*16 + l16); K split by lane-half
    FragU a;
    {
      const char* rp = sA + (w * 16 + l16) * 80;
      a.u[0] = *(const uint4*)(rp + half * 32);
      a.u[1] = *(const uint4*)(rp + half * 32 + 16);
    }
    #pragma unroll
    for (int c = 0; c < 8; ++c) {
      FragU b;
      const char* rp = sB + (c * 16 + l16) * 80;   // column n = c*16 + l16
      b.u[0] = *(const uint4*)(rp + half * 32);
      b.u[1] = *(const uint4*)(rp + half * 32 + 16);
      acc[c] = __builtin_amdgcn_wmma_f32_16x16x32_bf16(
          false, a.v, false, b.v, (short)0, acc[c], false, false);
    }

#if TDM_OK
    if (t < 32) tdm_wait0();
    wg_sync();
    buf = nxt;
#else
    wg_sync();
#endif
  }

  // C layout: row = r + 8*half, col = l16 (per 16-wide chunk c)
  #pragma unroll
  for (int c = 0; c < 8; ++c) {
    int col = nBase + c * 16 + l16;
    #pragma unroll
    for (int r = 0; r < 8; ++r) {
      int row = mBase + w * 16 + half * 8 + r;
      C[(size_t)row * N + col] = acc[c][r];
    }
  }
}

// ---------------------------------------------------------------------------
// RoPE + split QKV. qkv f32 [B*S, 6144] ->
//   Q bf16 [B,32,S,128], K bf16 [B,8,S,128], V^T bf16 [B,8,128,S]
// ---------------------------------------------------------------------------
__global__ void rope_split(const float* __restrict__ qkv,
                           const float* __restrict__ cosb,
                           const float* __restrict__ sinb,
                           unsigned short* __restrict__ qb,
                           unsigned short* __restrict__ kb,
                           unsigned short* __restrict__ vtb) {
  int i = blockIdx.x * blockDim.x + threadIdx.x;
  const int total = BB * SS * NQKV * (HD / 2);
  if (i >= total) return;
  int d2   = i & 63;
  int head = (i >> 6) % NQKV;
  int bs   = i / (NQKV * 64);
  int b = bs >> 10, s = bs & 1023;

  const float* p = qkv + (size_t)bs * QKVN + head * HD + 2 * d2;
  float t0 = p[0], t1 = p[1];
  float r0 = t0, r1 = t1;
  if (head < NH + NKV) {                       // rope on q and k only
    float c  = cosb[s * 64 + d2];
    float sn = sinb[s * 64 + d2];
    r0 = t0 * c - t1 * sn;
    r1 = t0 * sn + t1 * c;
  }
  if (head < NH) {
    unsigned short* o = qb + (((size_t)(b * NH + head)) * SS + s) * HD + 2 * d2;
    o[0] = f2bf(r0); o[1] = f2bf(r1);
  } else if (head < NH + NKV) {
    unsigned short* o = kb + (((size_t)(b * NKV + head - NH)) * SS + s) * HD + 2 * d2;
    o[0] = f2bf(r0); o[1] = f2bf(r1);
  } else {
    size_t base = ((size_t)(b * NKV + head - NH - NKV)) * HD;
    vtb[(base + 2 * d2) * SS + s]     = f2bf(r0);
    vtb[(base + 2 * d2 + 1) * SS + s] = f2bf(r1);
  }
}

// ---------------------------------------------------------------------------
// Flash attention (causal, GQA). Block = 128 Q rows of one (b,h); 8 waves x 16.
// KV chunk = 32.  K and V^T tiles staged by TDM.  Out: bf16 [B, S, 32*128]
// ---------------------------------------------------------------------------
__global__ __launch_bounds__(256) void attn_fa_wmma(
    const unsigned short* __restrict__ Q,
    const unsigned short* __restrict__ Kh,
    const unsigned short* __restrict__ Vt,
    unsigned short* __restrict__ Out) {
  __shared__ char smem[34816];
  lds_escape(smem);                      // TDM writes smem: make it visible
  const int t    = threadIdx.x;
  const int lane = t & 31;
  const int w    = t >> 5;
  const int half = lane >> 4;
  const int l16  = lane & 15;

  int blk  = blockIdx.x;
  int qblk = blk & 7;               // S/128 = 8
  int h    = (blk >> 3) & 31;
  int b    = blk >> 8;
  const int q0 = qblk * 128;
  const int hk = h >> 2;            // GROUPS = 4

  const unsigned short* qh = Q  + ((size_t)(b * NH  + h )) * SS * HD;
  const unsigned short* kh = Kh + ((size_t)(b * NKV + hk)) * SS * HD;
  const unsigned short* vh = Vt + ((size_t)(b * NKV + hk)) * HD * SS;

  // --- stage Q tile (128 x 128 bf16, row stride 272B), pull fragments ---
#if TDM_OK
  if (t < 32) {
    tdm_load_2d(0, qh + (size_t)q0 * HD, HD, 128, 128, 5, 3);  // 256B rows +16B pad
    tdm_wait0();
  }
#else
  {
    int row = t >> 1, ch = t & 1;
    const uint4* g = (const uint4*)(qh + (size_t)(q0 + row) * HD + ch * 64);
    uint4* s = (uint4*)(smem + row * 272 + ch * 128);
    #pragma unroll
    for (int i = 0; i < 8; ++i) s[i] = g[i];
  }
#endif
  wg_sync();
  FragU qf[4];
  {
    const char* rp = smem + (w * 16 + l16) * 272;
    #pragma unroll
    for (int c = 0; c < 4; ++c) {
      qf[c].u[0] = *(const uint4*)(rp + c * 64 + half * 32);
      qf[c].u[1] = *(const uint4*)(rp + c * 64 + half * 32 + 16);
    }
  }
  wg_sync();

  char* sK = smem;                         // 32 kv rows x 272B (128 bf16 + pad)
  char* sV = smem + 8704;                  // 128 hd rows x 80B (32 kv bf16 + pad)
  char* sP = smem + 18944 + w * 1280;      // per-wave P: 16 rows x 80B

  const v8f vzero = {0.f, 0.f, 0.f, 0.f, 0.f, 0.f, 0.f, 0.f};
  v8f o[8]; float m_r[8], l_r[8];
  #pragma unroll
  for (int c = 0; c < 8; ++c) o[c] = vzero;
  #pragma unroll
  for (int r = 0; r < 8; ++r) { m_r[r] = -1e30f; l_r[r] = 0.f; }

  const float scale = 0.08838834764831845f;   // 1/sqrt(128)
  const int kvEnd = q0 + 128;                  // causal upper bound for block

  for (int kv = 0; kv < kvEnd; kv += 32) {
#if TDM_OK
    if (t < 32) {
      tdm_load_2d(0,    kh + (size_t)kv * HD, HD,  128, 32, 5, 3);  // K chunk
      tdm_load_2d(8704, vh + kv,              SS,  32, 128, 3, 3);  // V^T chunk
      tdm_wait0();
    }
#else
    { // stage K chunk (row-major: contraction over HD is contiguous)
      int row = t >> 3, off = (t & 7) * 16;
      const uint4* g = (const uint4*)(kh + (size_t)(kv + row) * HD + off);
      uint4* s = (uint4*)(sK + row * 272 + off * 2);
      s[0] = g[0]; s[1] = g[1];
    }
    { // stage V^T chunk (rows = hd, contraction over kv contiguous)
      int row = t >> 1, off = (t & 1) * 16;
      const uint4* g = (const uint4*)(vh + (size_t)row * SS + kv + off);
      uint4* s = (uint4*)(sV + row * 80 + off * 2);
      s[0] = g[0]; s[1] = g[1];
    }
#endif
    wg_sync();

    // ---- scores: S = Q(16x128) @ K^T(128x32), two 16-wide N chunks ----
    v8f s0 = vzero, s1 = vzero;
    #pragma unroll
    for (int c = 0; c < 4; ++c) {
      FragU b0, b1;
      const char* r0p = sK + l16 * 272;
      const char* r1p = sK + (16 + l16) * 272;
      b0.u[0] = *(const uint4*)(r0p + c * 64 + half * 32);
      b0.u[1] = *(const uint4*)(r0p + c * 64 + half * 32 + 16);
      b1.u[0] = *(const uint4*)(r1p + c * 64 + half * 32);
      b1.u[1] = *(const uint4*)(r1p + c * 64 + half * 32 + 16);
      s0 = __builtin_amdgcn_wmma_f32_16x16x32_bf16(false, qf[c].v, false, b0.v,
                                                   (short)0, s0, false, false);
      s1 = __builtin_amdgcn_wmma_f32_16x16x32_bf16(false, qf[c].v, false, b1.v,
                                                   (short)0, s1, false, false);
    }

    // ---- online softmax (row = q0 + 16w + 8*half + r, col = l16) ----
    const int qrow_base = q0 + w * 16 + half * 8;
    #pragma unroll
    for (int r = 0; r < 8; ++r) {
      int qr = qrow_base + r;
      float a0 = s0[r] * scale; if (kv + l16 > qr)      a0 = -1e9f;
      float a1 = s1[r] * scale; if (kv + 16 + l16 > qr) a1 = -1e9f;
      float cm = rmax16(fmaxf(a0, a1));
      float mn = fmaxf(m_r[r], cm);
      float sc = __expf(m_r[r] - mn);
      float p0 = __expf(a0 - mn);
      float p1 = __expf(a1 - mn);
      float rs = rsum16(p0 + p1);
      l_r[r] = l_r[r] * sc + rs;
      m_r[r] = mn;
      #pragma unroll
      for (int c = 0; c < 8; ++c) o[c][r] = o[c][r] * sc;
      unsigned short* pp = (unsigned short*)(sP + (half * 8 + r) * 80);
      pp[l16]      = f2bf(p0);
      pp[16 + l16] = f2bf(p1);
    }
    wg_sync();

    // ---- O += P(16x32) @ V(32x128) ----
    FragU pf;
    pf.u[0] = *(const uint4*)(sP + l16 * 80 + half * 32);
    pf.u[1] = *(const uint4*)(sP + l16 * 80 + half * 32 + 16);
    #pragma unroll
    for (int c = 0; c < 8; ++c) {
      FragU vf;
      const char* rp = sV + (c * 16 + l16) * 80;
      vf.u[0] = *(const uint4*)(rp + half * 32);
      vf.u[1] = *(const uint4*)(rp + half * 32 + 16);
      o[c] = __builtin_amdgcn_wmma_f32_16x16x32_bf16(false, pf.v, false, vf.v,
                                                     (short)0, o[c], false, false);
    }
    wg_sync();
  }

  // ---- normalize + store bf16 [b, q, h*128 + hd] ----
  #pragma unroll
  for (int r = 0; r < 8; ++r) {
    int qr = q0 + w * 16 + half * 8 + r;
    float inv = 1.f / l_r[r];
    size_t base = ((size_t)(b * SS + qr)) * (NH * HD) + h * HD;
    #pragma unroll
    for (int c = 0; c < 8; ++c)
      Out[base + c * 16 + l16] = f2bf(o[c][r] * inv);
  }
}

// ---------------------------------------------------------------------------
// Host-side launch
// ---------------------------------------------------------------------------
extern "C" void kernel_launch(void* const* d_in, const int* in_sizes, int n_in,
                              void* d_out, int out_size, void* d_ws, size_t ws_size,
                              hipStream_t stream) {
  const float* x     = (const float*)d_in[0];   // [B,S,H]
  const float* w_qkv = (const float*)d_in[1];   // [H, 6144]
  const float* w_o   = (const float*)d_in[2];   // [4096, H]
  const float* cosb  = (const float*)d_in[3];   // [S, 64]
  const float* sinb  = (const float*)d_in[4];   // [S, 64]
  // d_in[5] (mask) unused: causal mask applied analytically.
  float* out = (float*)d_out;

  char* ws = (char*)d_ws;
  unsigned short* x_bf     = (unsigned short*)(ws);                  // 16 MB
  unsigned short* wqkvT_bf = (unsigned short*)(ws + 16777216);       // 48 MB  [6144,4096]
  unsigned short* woT_bf   = (unsigned short*)(ws + 67108864);       // 32 MB  [4096,4096]
  float*          qkv_f    = (float*)         (ws + 100663296);      // 48 MB
  unsigned short* q_bf     = (unsigned short*)(ws + 150994944);      // 16 MB
  unsigned short* k_bf     = (unsigned short*)(ws + 167772160);      //  4 MB
  unsigned short* vt_bf    = (unsigned short*)(ws + 171966464);      //  4 MB
  unsigned short* attn_bf  = (unsigned short*)(ws + 176160768);      // 16 MB

  const int nx = BB * SS * HH;        // 8388608

  cvt_f32_bf16<<<(nx + 255) / 256, 256, 0, stream>>>(x, x_bf, nx);
  // weights: transpose + convert so GEMM B tiles are contraction-contiguous
  transpose_cvt<<<dim3(QKVN / 32, HH / 32), 256, 0, stream>>>(w_qkv, wqkvT_bf, HH, QKVN);
  transpose_cvt<<<dim3(HH / 32, HH / 32), 256, 0, stream>>>(w_o, woT_bf, HH, HH);

  // QKV projection: [2048,4096] @ [4096,6144] -> f32
  gemm_bf16_wmma<<<dim3(QKVN / 128, (BB * SS) / 128), 256, 0, stream>>>(
      x_bf, wqkvT_bf, qkv_f, BB * SS, QKVN, HH);

  // RoPE + head split (+ V transpose)
  {
    int total = BB * SS * NQKV * (HD / 2);
    rope_split<<<(total + 255) / 256, 256, 0, stream>>>(qkv_f, cosb, sinb,
                                                        q_bf, k_bf, vt_bf);
  }

  // Causal GQA flash attention
  attn_fa_wmma<<<BB * NH * (SS / 128), 256, 0, stream>>>(q_bf, k_bf, vt_bf,
                                                         attn_bf);

  // Output projection: [2048,4096] @ [4096,4096] -> f32 out
  gemm_bf16_wmma<<<dim3(HH / 128, (BB * SS) / 128), 256, 0, stream>>>(
      attn_bf, woT_bf, out, BB * SS, HH, HH);
}
